// FrugalRnn_14645838479498
// MI455X (gfx1250) — compile-verified, matrix operand order checked by reading, not run on hardware
//
#include <hip/hip_runtime.h>
#include <hip/hip_bf16.h>

// ---------------------------------------------------------------------------
// FrugalRnn (GRU ponder loop) for gfx1250 / MI455X, wave32 + WMMA bf16.
//   B=65536, I=128, H=256, BUDGET=16.
// v3:
//   * 16 batch rows per 512-thread WG (1 M-tile per wave) to keep per-wave
//     VGPRs ~270 (B-operands 192 + acc 24 + gi 12 + misc) -> no spills while
//     still hoisting ALL global operands out of the iteration loop.
//   * gi stored in native WMMA C-tile layout [mt][nt][lane][v].
//   * Steady-state loop: LDS A-loads + 24 v_wmma + fused GRU epilogue only.
// ---------------------------------------------------------------------------

typedef __bf16 bf16_t;
typedef __bf16 v16bf __attribute__((ext_vector_type(16)));
typedef __bf16 v8bf  __attribute__((ext_vector_type(8)));
typedef float  v8f   __attribute__((ext_vector_type(8)));

#define BN   65536
#define IN_  128
#define HN   256
#define G3   768      // 3*H
#define BUD  16
#define NT3  48       // 768/16 column tiles

// ---------------------------------------------------------------------------
// A-operand loader for V_WMMA_F32_16X16X32_BF16 (16-bit A, 16x32):
//   lanes 0-15  hold row M=lane,    K = [0..7]  (v0-3) and [16..23] (v4-7)
//   lanes 16-31 hold row M=lane-16, K = [8..15] (v0-3) and [24..31] (v4-7)
// Caller passes pointer to (row, k0 + half*8); we read 8 + 8 contiguous bf16.
// ---------------------------------------------------------------------------
static __device__ __forceinline__ v16bf load_a16(const bf16_t* p) {
    v16bf a;
#pragma unroll
    for (int i = 0; i < 8; ++i) a[i] = p[i];
#pragma unroll
    for (int i = 0; i < 8; ++i) a[8 + i] = p[16 + i];
    return a;
}

static __device__ __forceinline__ float sigmoidf_(float x) {
    return 1.0f / (1.0f + __expf(-x));
}

// ---------------------------------------------------------------------------
// Pack a row-major [N,K] f32 weight into bf16 B-operand tiles: for tile
// (nt, ks) each lane owns 16 contiguous bf16 (32 B) -> one aligned 32-byte
// vector load per WMMA. B[k][n] = w[n][k] (we need w^T).
//   lanes 0-15:  n = nt*16+lane,    K = ks*32 + [0..7] then [16..23]
//   lanes 16-31: n = nt*16+lane-16, K = ks*32 + [8..15] then [24..31]
// ---------------------------------------------------------------------------
__global__ __launch_bounds__(256) void pack_w(const float* __restrict__ w,
                                              bf16_t* __restrict__ out,
                                              int N, int K) {
    int idx = blockIdx.x * 256 + threadIdx.x;
    int total = N * K;
    if (idx >= total) return;
    int ksteps = K >> 5;
    int j    = idx & 15;
    int l    = (idx >> 4) & 31;
    int rest = idx >> 9;
    int ks   = rest % ksteps;
    int nt   = rest / ksteps;
    int n     = nt * 16 + (l & 15);
    int kbase = ks * 32 + ((l >> 4) ? 8 : 0);
    int k     = (j < 8) ? (kbase + j) : (kbase + 16 + (j - 8));
    out[idx] = (bf16_t)w[(size_t)n * K + k];
}

__global__ __launch_bounds__(256) void cvt_bf16(const float* __restrict__ in,
                                                bf16_t* __restrict__ out,
                                                int n) {
    int i = blockIdx.x * 256 + threadIdx.x;
    if (i < n) out[i] = (bf16_t)in[i];
}

// ---------------------------------------------------------------------------
// gi = x @ w_ih^T + b_ih -> bf16, stored in WMMA C-tile layout:
//   gi[((mt*48 + nt)*256) + lane*8 + v]  == C tile (mt,nt) element (lane,v)
// One 16x16 output tile per wave; epilogue is a single b128 store per lane.
// ---------------------------------------------------------------------------
__global__ __launch_bounds__(256)
void gi_gemm(const bf16_t* __restrict__ xb, const bf16_t* __restrict__ wp,
             const float* __restrict__ b_ih, bf16_t* __restrict__ gi) {
    const int wave = blockIdx.x * 8 + (threadIdx.x >> 5);
    const int lane = threadIdx.x & 31;
    const int nt   = wave % NT3;
    const int mt   = wave / NT3;
    const int m    = lane & 15;
    const int half = lane >> 4;

    v8f acc = {};
#pragma unroll
    for (int ks = 0; ks < 4; ++ks) {   // K = 128 = 4 * 32
        const bf16_t* ap = xb + (size_t)(mt * 16 + m) * IN_ + ks * 32 + half * 8;
        v16bf a = load_a16(ap);
        v16bf b = *(const v16bf*)(wp + (((size_t)nt * 4 + ks) * 32 + lane) * 16);
        acc = __builtin_amdgcn_wmma_f32_16x16x32_bf16(false, a, false, b,
                                                      (short)0, acc, false, false);
    }
    // C layout: lanes 0-15 -> M = v, N = lane; lanes 16-31 -> M = v+8.
    const float bi = b_ih[nt * 16 + m];
    v8bf st;
#pragma unroll
    for (int v = 0; v < 8; ++v) st[v] = (bf16_t)(acc[v] + bi);
    *(v8bf*)(gi + ((size_t)(mt * NT3 + nt) * 256 + lane * 8)) = st;
}

// ---------------------------------------------------------------------------
// Persistent GRU ponder loop. 512 threads = 16 waves; 16 batch rows per WG.
// Wave ct owns gate-column tile ct (16 h-columns) for the single 16-row
// M-tile, accumulating the r/z/n gate tiles (N-tiles ct, ct+16, ct+32 of the
// 768-wide gh) so the GRU nonlinearity fuses into the WMMA epilogue.
// All global operands (w_hh B-tiles, gi C-tiles, head weights) are hoisted
// into registers/LDS before the loop: the steady state is LDS + WMMA only.
// ---------------------------------------------------------------------------
__global__ __launch_bounds__(512)
void gru_persist(const bf16_t* __restrict__ gi, const bf16_t* __restrict__ whh,
                 const float* __restrict__ b_hh,
                 const float* __restrict__ w_stop, const float* __restrict__ b_stop,
                 const float* __restrict__ w_pred, const float* __restrict__ b_pred,
                 float* __restrict__ out) {
    __shared__ bf16_t   hs[16][HN];     // hidden state, bf16 (8 KB)
    __shared__ float    red_s[16][32];
    __shared__ float    red_p[16][32];
    __shared__ float    wsv[HN];        // stop/pred head weights (staged once)
    __shared__ float    wpv[HN];
    __shared__ float    s_prob[16];
    __shared__ float    s_iters[16];
    __shared__ int      s_halt[16];
    __shared__ unsigned s_hmask;
    __shared__ int      s_active;

    const int tid  = threadIdx.x;
    const int wave = tid >> 5;        // 0..15 == gate-column tile ct
    const int lane = tid & 31;
    const int ct   = wave;
    const int m16  = lane & 15;
    const int half = lane >> 4;
    const int mt   = blockIdx.x;      // this WG's 16-row M-tile
    const size_t rowBase = (size_t)mt * 16;

    // ---- init hidden (h = 0 except h[:,0] = BUDGET) and per-row state ----
    for (int idx = tid; idx < 16 * HN; idx += 512) {
        int r = idx >> 8, c = idx & (HN - 1);
        hs[r][c] = (bf16_t)((c == 0) ? 16.0f : 0.0f);
    }
    for (int idx = tid; idx < HN; idx += 512) { wsv[idx] = w_stop[idx]; wpv[idx] = w_pred[idx]; }
    if (tid < 16) { s_prob[tid] = 0.0f; s_iters[tid] = 0.0f; s_halt[tid] = 0; }

    // ---- hoist loop-invariant operands into registers ----
    // w_hh B-operand slices for this wave's three gate N-tiles (192 VGPRs)
    v16bf bR[8], bZ[8], bN[8];
#pragma unroll
    for (int ks = 0; ks < 8; ++ks) {
        bR[ks] = *(const v16bf*)(whh + (((size_t)(ct     ) * 8 + ks) * 32 + lane) * 16);
        bZ[ks] = *(const v16bf*)(whh + (((size_t)(ct + 16) * 8 + ks) * 32 + lane) * 16);
        bN[ks] = *(const v16bf*)(whh + (((size_t)(ct + 32) * 8 + ks) * 32 + lane) * 16);
    }
    // gi C-tiles in matching (lane,v) layout: 3 x b128 = 12 VGPRs
    v8bf giR = *(const v8bf*)(gi + ((size_t)(mt * NT3 + ct     ) * 256 + lane * 8));
    v8bf giZ = *(const v8bf*)(gi + ((size_t)(mt * NT3 + ct + 16) * 256 + lane * 8));
    v8bf giN = *(const v8bf*)(gi + ((size_t)(mt * NT3 + ct + 32) * 256 + lane * 8));

    // private f32 copy of this lane's own hidden elements (rows half*8+v,
    // column c) -- exactly the elements this lane commits each iteration.
    const int   c     = ct * 16 + m16;
    const float hinit = (c == 0) ? 16.0f : 0.0f;
    float hreg[8];
#pragma unroll
    for (int v = 0; v < 8; ++v) hreg[v] = hinit;

    const float bh_r = b_hh[c];
    const float bh_z = b_hh[HN + c];
    const float bh_n = b_hh[2 * HN + c];

    __syncthreads();

    for (int it = 0; it < BUD; ++it) {
        if (tid == 0) {
            unsigned m = 0; int a = 0;
            for (int r = 0; r < 16; ++r) { if (s_halt[r]) m |= 1u << r; else ++a; }
            s_hmask = m; s_active = a;
        }
        __syncthreads();
        const unsigned hmask = s_hmask;
        if (s_active == 0) break;     // uniform: whole tile halted

        // ---- gh = h @ w_hh^T : 3 gate N-tiles, K = 256 ----
        v8f aR = {}, aZ = {}, aN = {};
#pragma unroll
        for (int ks = 0; ks < 8; ++ks) {
            const int k0 = ks * 32 + half * 8;
            v16bf A = load_a16(&hs[m16][k0]);
            aR = __builtin_amdgcn_wmma_f32_16x16x32_bf16(false, A, false, bR[ks], (short)0, aR, false, false);
            aZ = __builtin_amdgcn_wmma_f32_16x16x32_bf16(false, A, false, bZ[ks], (short)0, aZ, false, false);
            aN = __builtin_amdgcn_wmma_f32_16x16x32_bf16(false, A, false, bN[ks], (short)0, aN, false, false);
        }

        // ---- fused GRU gate epilogue (all operands in registers) ----
        float nh[8];
#pragma unroll
        for (int v = 0; v < 8; ++v) {
            float rr = sigmoidf_((float)giR[v] + aR[v] + bh_r);
            float zz = sigmoidf_((float)giZ[v] + aZ[v] + bh_z);
            float nn = tanhf((float)giN[v] + rr * (aN[v] + bh_n));
            nh[v] = (1.0f - zz) * nn + zz * hreg[v];
        }
        __syncthreads();   // all waves done reading hs

        // commit new hidden only for still-active rows (masked, per ref)
#pragma unroll
        for (int v = 0; v < 8; ++v) {
            int r = half * 8 + v;
            if (!((hmask >> r) & 1u)) { hs[r][c] = (bf16_t)nh[v]; hreg[v] = nh[v]; }
        }
        __syncthreads();

        // ---- stop / pred heads: 32 threads per row, LDS reduction ----
        {
            int r = tid >> 5, sub = tid & 31;
            float ss = 0.0f, sp = 0.0f;
            int cb = sub * 8;
#pragma unroll
            for (int k = 0; k < 8; ++k) {
                float hv = (float)hs[r][cb + k];
                ss += hv * wsv[cb + k];
                sp += hv * wpv[cb + k];
            }
            red_s[r][sub] = ss;
            red_p[r][sub] = sp;
        }
        __syncthreads();
        if (tid < 16) {
            int r = tid;
            float ss = 0.0f, sp = 0.0f;
#pragma unroll
            for (int s2 = 0; s2 < 32; ++s2) { ss += red_s[r][s2]; sp += red_p[r][s2]; }
            ss += b_stop[0];
            sp += b_pred[0];
            int haltNow = (sigmoidf_(ss) >= 0.5f);
            if (s_halt[r] == 0) {                     // active
                s_iters[r] += 1.0f;
                if (haltNow || it == BUD - 1) s_prob[r] = sp;
                if (haltNow) s_halt[r] = 1;
            }
        }
        __syncthreads();
    }

    if (tid < 16) {
        size_t g = rowBase + tid;
        out[g]              = s_prob[tid];   // final_probs
        out[(size_t)BN + g] = s_iters[tid];  // n_iters
    }
}

// ---------------------------------------------------------------------------
extern "C" void kernel_launch(void* const* d_in, const int* in_sizes, int n_in,
                              void* d_out, int out_size, void* d_ws, size_t ws_size,
                              hipStream_t stream) {
    const float* x      = (const float*)d_in[0];
    const float* w_ih   = (const float*)d_in[1];
    const float* w_hh   = (const float*)d_in[2];
    const float* b_ih   = (const float*)d_in[3];
    const float* b_hh   = (const float*)d_in[4];
    const float* w_stop = (const float*)d_in[5];
    const float* b_stop = (const float*)d_in[6];
    const float* w_pred = (const float*)d_in[7];
    const float* b_pred = (const float*)d_in[8];
    float* out = (float*)d_out;

    // Workspace layout (bf16), all offsets 256B-aligned:
    //   xb   : B*I     = 16.8 MB   (x in bf16)
    //   wihp : 768*128 = 196 KB    (packed w_ih B-operand)
    //   whhp : 768*256 = 393 KB    (packed w_hh B-operand, L2-resident)
    //   gib  : B*768   = 100.7 MB  (loop-invariant gi, WMMA C-tile layout)
    char* ws = (char*)d_ws;
    size_t off = 0;
    bf16_t* xb   = (bf16_t*)(ws + off); off += (size_t)BN * IN_ * 2;
    bf16_t* wihp = (bf16_t*)(ws + off); off += (size_t)G3 * IN_ * 2;
    bf16_t* whhp = (bf16_t*)(ws + off); off += (size_t)G3 * HN * 2;
    bf16_t* gib  = (bf16_t*)(ws + off);

    cvt_bf16<<<(BN * IN_) / 256, 256, 0, stream>>>(x, xb, BN * IN_);
    pack_w<<<(G3 * IN_ + 255) / 256, 256, 0, stream>>>(w_ih, wihp, G3, IN_);
    pack_w<<<(G3 * HN + 255) / 256, 256, 0, stream>>>(w_hh, whhp, G3, HN);

    // gi GEMM: (B/16)*(768/16) = 196608 wave-tiles, 8 waves/block
    gi_gemm<<<(BN / 16) * NT3 / 8, 256, 0, stream>>>(xb, wihp, b_ih, gib);

    // Persistent GRU loop: 16 rows per 512-thread workgroup
    gru_persist<<<BN / 16, 512, 0, stream>>>(gib, whhp, b_hh,
                                             w_stop, b_stop, w_pred, b_pred, out);
}
